// MultiHeadAttention_53910429499736
// MI455X (gfx1250) — matrix operand order repaired and checked by reading, not compiled, not run on hardware
//
#include <hip/hip_runtime.h>
#include <hip/hip_bf16.h>

typedef __attribute__((ext_vector_type(16))) _Float16 v16h;
typedef __attribute__((ext_vector_type(8)))  _Float16 v8h;
typedef __attribute__((ext_vector_type(4)))  _Float16 v4h;
typedef __attribute__((ext_vector_type(8)))  float    v8f;

#define WMMA_F16(A, B, C) \
  __builtin_amdgcn_wmma_f32_16x16x32_f16(false, (A), false, (B), (short)0, (C), false, false)

static constexpr int BB   = 4;
static constexpr int SS   = 2048;
static constexpr int DD   = 1024;
static constexpr int HH   = 16;
static constexpr int HD   = 64;
static constexpr int MTOT = BB * SS;        // 8192 token rows

// ---------------------------------------------------------------------------
// Kernel 1: fp32 -> f16 convert of x (4 elements / thread, vectorized)
// ---------------------------------------------------------------------------
__global__ __launch_bounds__(256) void cvt_x(const float4* __restrict__ in,
                                             _Float16* __restrict__ out) {
  int i = blockIdx.x * blockDim.x + threadIdx.x;
  float4 v = in[i];
  v4h h;
  h[0] = (_Float16)v.x; h[1] = (_Float16)v.y;
  h[2] = (_Float16)v.z; h[3] = (_Float16)v.w;
  *(v4h*)(out + (size_t)i * 4) = h;
}

// ---------------------------------------------------------------------------
// Kernel 2: tiled fp32 -> f16 transpose: in[R][C] -> out[C][R]
// ---------------------------------------------------------------------------
__global__ __launch_bounds__(256) void transpose_cvt(const float* __restrict__ in,
                                                     _Float16* __restrict__ out,
                                                     int R, int C) {
  __shared__ float tile[32][33];
  int c0 = blockIdx.x * 32, r0 = blockIdx.y * 32;
  int tx = threadIdx.x & 31, ty = threadIdx.x >> 5;   // ty in 0..7
  #pragma unroll
  for (int i = ty; i < 32; i += 8)
    tile[i][tx] = in[(size_t)(r0 + i) * C + c0 + tx];
  __syncthreads();
  #pragma unroll
  for (int i = ty; i < 32; i += 8)
    out[(size_t)(c0 + i) * R + r0 + tx] = (_Float16)tile[tx][i];
}

// ---------------------------------------------------------------------------
// A-fragment loader: row-major f16 source, 16x32 tile at (row base, col kb)
// lane: M = lane&15, half = lane>>4 ; element e -> K = half*16 + e
// ---------------------------------------------------------------------------
__device__ __forceinline__ v16h load_a_frag(const _Float16* __restrict__ row_base,
                                            int row_stride, int kb, int m, int hh) {
  const _Float16* p = row_base + (size_t)m * row_stride + kb + hh * 16;
  v8h lo = *(const v8h*)(p);
  v8h hi = *(const v8h*)(p + 8);
  v16h a;
  #pragma unroll
  for (int i = 0; i < 8; ++i) { a[i] = lo[i]; a[8 + i] = hi[i]; }
  return a;
}

// ---------------------------------------------------------------------------
// Kernel 3: QKV GEMM. xh[8192][1024] @ wqkvT[3072][1024]^T -> q/k/vT buffers.
// One wave = 32x64 output tile (8 WMMAs / k-step, B reused 2x, A reused 4x).
// tasks = 256 mtiles * 48 ntiles = 12288 waves -> 1536 blocks.
// ---------------------------------------------------------------------------
__global__ __launch_bounds__(256) void qkv_gemm(const _Float16* __restrict__ xh,
                                                const _Float16* __restrict__ wT,
                                                _Float16* __restrict__ qb,
                                                _Float16* __restrict__ kbuf,
                                                _Float16* __restrict__ vt) {
  int wave = threadIdx.x >> 5, lane = threadIdx.x & 31;
  int task  = blockIdx.x * 8 + wave;
  int mt    = task / 48, nt64 = task % 48;
  int mbase = mt * 32,  nbase = nt64 * 64;
  int m = lane & 15, hh = lane >> 4;

  const _Float16* arow0 = xh + (size_t)mbase * DD;
  const _Float16* arow1 = arow0 + (size_t)16 * DD;
  v8f acc[2][4] = {};

  for (int kb = 0; kb < DD; kb += 32) {
    v16h a0 = load_a_frag(arow0, DD, kb, m, hh);
    v16h a1 = load_a_frag(arow1, DD, kb, m, hh);
    #pragma unroll
    for (int nt = 0; nt < 4; ++nt) {
      const _Float16* bp = wT + (size_t)(nbase + nt * 16 + m) * DD + kb + hh * 16;
      v16h b = *(const v16h*)bp;
      acc[0][nt] = WMMA_F16(a0, b, acc[0][nt]);
      acc[1][nt] = WMMA_F16(a1, b, acc[1][nt]);
    }
  }

  #pragma unroll
  for (int nt = 0; nt < 4; ++nt) {
    int c     = nbase + nt * 16;       // tile column base (16-aligned)
    int which = c >> 10;               // 0=q 1=k 2=v (uniform per tile)
    int rem   = c & 1023;
    int head  = rem >> 6;              // uniform per tile
    int dim   = (rem & 63) + m;        // per-lane dim within head
    #pragma unroll
    for (int mi = 0; mi < 2; ++mi) {
      #pragma unroll
      for (int r = 0; r < 8; ++r) {
        int s    = mbase + mi * 16 + r + 8 * hh;   // token index
        int bidx = s >> 11, srow = s & (SS - 1);
        _Float16 val = (_Float16)acc[mi][nt][r];
        size_t bh = (size_t)(bidx * HH + head);
        if (which == 0)      qb  [(bh * SS + srow) * HD + dim] = val;
        else if (which == 1) kbuf[(bh * SS + srow) * HD + dim] = val;
        else                 vt  [(bh * HD + dim) * SS + srow] = val;  // V^T
      }
    }
  }
}

// ---------------------------------------------------------------------------
// Kernel 4: flash attention. One wave = (b,h, 32-query tile): two 16-row
// M-subtiles sharing every K/V fragment (halves K/V traffic). Row-sum of P is
// computed by the matrix unit via an all-ones B fragment (no shuffle-reduce).
// tasks = 4*16*64 = 4096 waves -> 512 blocks.
// ---------------------------------------------------------------------------
__global__ __launch_bounds__(256) void attn_kernel(const _Float16* __restrict__ q,
                                                   const _Float16* __restrict__ k,
                                                   const _Float16* __restrict__ vt,
                                                   _Float16* __restrict__ attn_out) {
  __shared__ _Float16 plds[8][2][512];               // 2KB per wave (2x 16x32 f16)
  int wave = threadIdx.x >> 5, lane = threadIdx.x & 31;
  int task = blockIdx.x * 8 + wave;                  // (b*16+h)*64 + qt
  int bh   = task >> 6;                              // b*16 + h
  int qt   = task & 63;                              // 32-query tile index
  int m = lane & 15, hh = lane >> 4;

  const _Float16* Q = q  + ((size_t)bh * SS + qt * 32) * HD;
  const _Float16* K = k  + (size_t)bh * SS * HD;
  const _Float16* V = vt + (size_t)bh * HD * SS;

  // Q A-fragments [tile][kchunk], pre-scaled by 1/sqrt(64) = 0.125
  v16h aq[2][2];
  const _Float16 sc = (_Float16)0.125f;
  #pragma unroll
  for (int t = 0; t < 2; ++t) {
    const _Float16* qp = Q + (size_t)(t * 16 + m) * HD + hh * 16;
    v8h c0 = *(const v8h*)(qp);      v8h c1 = *(const v8h*)(qp + 8);
    v8h c2 = *(const v8h*)(qp + 32); v8h c3 = *(const v8h*)(qp + 40);
    #pragma unroll
    for (int i = 0; i < 8; ++i) {
      aq[t][0][i] = c0[i] * sc; aq[t][0][8 + i] = c1[i] * sc;
      aq[t][1][i] = c2[i] * sc; aq[t][1][8 + i] = c3[i] * sc;
    }
  }

  // all-ones B fragment: used to compute row-sums of P on the matrix unit
  v16h ones;
  #pragma unroll
  for (int i = 0; i < 16; ++i) ones[i] = (_Float16)1.0f;

  v8f o[2][4] = {};          // output accumulators
  v8f oL[2] = {};            // row-sum accumulators (l), via P @ ones
  float mr[2][8];
  #pragma unroll
  for (int t = 0; t < 2; ++t)
    #pragma unroll
    for (int r = 0; r < 8; ++r) mr[t][r] = -1e30f;

  #pragma unroll 1
  for (int j0 = 0; j0 < SS; j0 += 32) {
    // ---- shared K fragments for this 32-key block ----
    const _Float16* kp0 = K + (size_t)(j0 + m) * HD + hh * 16;
    const _Float16* kp1 = K + (size_t)(j0 + 16 + m) * HD + hh * 16;
    __builtin_prefetch((const void*)(kp0 + 32 * HD), 0, 0);   // next key block
    v16h bk0a = *(const v16h*)(kp0);
    v16h bk0b = *(const v16h*)(kp0 + 32);
    v16h bk1a = *(const v16h*)(kp1);
    v16h bk1b = *(const v16h*)(kp1 + 32);

    #pragma unroll
    for (int t = 0; t < 2; ++t) {
      // logits for 16 queries x 32 keys
      v8f s0 = {}, s1 = {};
      s0 = WMMA_F16(aq[t][0], bk0a, s0);
      s0 = WMMA_F16(aq[t][1], bk0b, s0);
      s1 = WMMA_F16(aq[t][0], bk1a, s1);
      s1 = WMMA_F16(aq[t][1], bk1b, s1);

      // row max across the 16-lane N group
      float bm[8];
      #pragma unroll
      for (int r = 0; r < 8; ++r) bm[r] = fmaxf(s0[r], s1[r]);
      #pragma unroll
      for (int off = 1; off < 16; off <<= 1)
        #pragma unroll
        for (int r = 0; r < 8; ++r) bm[r] = fmaxf(bm[r], __shfl_xor(bm[r], off, 32));

      float p0[8], p1[8];
      #pragma unroll
      for (int r = 0; r < 8; ++r) {
        float mn = fmaxf(mr[t][r], bm[r]);
        float al = __expf(mr[t][r] - mn);
        mr[t][r] = mn;
        p0[r] = __expf(s0[r] - mn);
        p1[r] = __expf(s1[r] - mn);
        o[t][0][r] *= al; o[t][1][r] *= al;
        o[t][2][r] *= al; o[t][3][r] *= al;
        oL[t][r]   *= al;
      }
      // store P^T into this wave's LDS slab (contiguous 8-half chunks)
      v8h ph0, ph1;
      #pragma unroll
      for (int r = 0; r < 8; ++r) { ph0[r] = (_Float16)p0[r]; ph1[r] = (_Float16)p1[r]; }
      *(v8h*)(&plds[wave][t][(m)      * 16 + hh * 8]) = ph0;
      *(v8h*)(&plds[wave][t][(16 + m) * 16 + hh * 8]) = ph1;
    }
    __syncthreads();
    v16h pa[2];
    #pragma unroll
    for (int t = 0; t < 2; ++t)
      #pragma unroll
      for (int e = 0; e < 16; ++e) {
        int g = e >> 3, j = e & 7;
        pa[t][e] = plds[wave][t][(hh * 16 + g * 8 + j) * 16 + m];
      }
    __syncthreads();

    // ---- shared V fragments; o += P @ V ; oL += P @ ones ----
    const _Float16* vb = V + (size_t)m * SS + j0 + hh * 16;
    __builtin_prefetch((const void*)(vb + 32), 0, 0);
    v16h bv0 = *(const v16h*)(vb);
    v16h bv1 = *(const v16h*)(vb + (size_t)16 * SS);
    v16h bv2 = *(const v16h*)(vb + (size_t)32 * SS);
    v16h bv3 = *(const v16h*)(vb + (size_t)48 * SS);
    #pragma unroll
    for (int t = 0; t < 2; ++t) {
      o[t][0] = WMMA_F16(pa[t], bv0, o[t][0]);
      o[t][1] = WMMA_F16(pa[t], bv1, o[t][1]);
      o[t][2] = WMMA_F16(pa[t], bv2, o[t][2]);
      o[t][3] = WMMA_F16(pa[t], bv3, o[t][3]);
      oL[t]   = WMMA_F16(pa[t], ones, oL[t]);   // row-sum on the matrix unit
    }
  }

  // ---- epilogue: normalize and write attn_out [B*S][1024] (f16) ----
  int bidx = bh >> 4, head = bh & 15;
  #pragma unroll
  for (int t = 0; t < 2; ++t) {
    #pragma unroll
    for (int r = 0; r < 8; ++r) {
      float inv = 1.0f / oL[t][r];   // every lane in the N-group holds row's l
      size_t row = (size_t)bidx * SS + qt * 32 + t * 16 + r + 8 * hh;
      _Float16* dst = attn_out + row * DD + head * HD + m;
      dst[0]  = (_Float16)(o[t][0][r] * inv);
      dst[16] = (_Float16)(o[t][1][r] * inv);
      dst[32] = (_Float16)(o[t][2][r] * inv);
      dst[48] = (_Float16)(o[t][3][r] * inv);
    }
  }
}

// ---------------------------------------------------------------------------
// Kernel 5: output projection. attn[8192][1024] @ woutT[1024][1024]^T -> fp32.
// One wave = 32x64 tile. tasks = 256 * 16 = 4096 waves -> 512 blocks.
// ---------------------------------------------------------------------------
__global__ __launch_bounds__(256) void out_gemm(const _Float16* __restrict__ ah,
                                                const _Float16* __restrict__ wT,
                                                float* __restrict__ out) {
  int wave = threadIdx.x >> 5, lane = threadIdx.x & 31;
  int task  = blockIdx.x * 8 + wave;
  int mt    = task >> 4, nt64 = task & 15;
  int mbase = mt * 32,   nbase = nt64 * 64;
  int m = lane & 15, hh = lane >> 4;

  const _Float16* arow0 = ah + (size_t)mbase * DD;
  const _Float16* arow1 = arow0 + (size_t)16 * DD;
  v8f acc[2][4] = {};

  for (int kb = 0; kb < DD; kb += 32) {
    v16h a0 = load_a_frag(arow0, DD, kb, m, hh);
    v16h a1 = load_a_frag(arow1, DD, kb, m, hh);
    #pragma unroll
    for (int nt = 0; nt < 4; ++nt) {
      const _Float16* bp = wT + (size_t)(nbase + nt * 16 + m) * DD + kb + hh * 16;
      v16h b = *(const v16h*)bp;
      acc[0][nt] = WMMA_F16(a0, b, acc[0][nt]);
      acc[1][nt] = WMMA_F16(a1, b, acc[1][nt]);
    }
  }
  #pragma unroll
  for (int mi = 0; mi < 2; ++mi)
    #pragma unroll
    for (int nt = 0; nt < 4; ++nt)
      #pragma unroll
      for (int r = 0; r < 8; ++r)
        out[(size_t)(mbase + mi * 16 + r + 8 * hh) * DD + nbase + nt * 16 + m] =
            acc[mi][nt][r];
}

// ---------------------------------------------------------------------------
// Host-side launcher
// ---------------------------------------------------------------------------
extern "C" void kernel_launch(void* const* d_in, const int* in_sizes, int n_in,
                              void* d_out, int out_size, void* d_ws, size_t ws_size,
                              hipStream_t stream) {
  const float* x      = (const float*)d_in[0];   // [4,2048,1024]
  const float* w_qkv  = (const float*)d_in[1];   // [1024,3072]
  const float* w_out  = (const float*)d_in[2];   // [1024,1024]
  float*       out    = (float*)d_out;           // [4,2048,1024]

  char* ws = (char*)d_ws;
  // workspace layout (bytes)
  _Float16* xh    = (_Float16*)(ws);                      // 16 MB  : x in f16
  _Float16* wqkvT = (_Float16*)(ws + 16777216);           //  6 MB  : w_qkv^T f16
  _Float16* woutT = (_Float16*)(ws + 23068672);           //  2 MB  : w_out^T f16
  _Float16* qb    = (_Float16*)(ws + 25165824);           // 16 MB  : Q  [B,H,S,64]
  _Float16* kbuf  = (_Float16*)(ws + 41943040);           // 16 MB  : K  [B,H,S,64]
  _Float16* vt    = (_Float16*)(ws + 58720256);           // 16 MB  : V^T[B,H,64,S]
  _Float16* attn  = (_Float16*)(ws + 75497472);           // 16 MB  : attn [B*S,1024]

  // 1) converts / transposes
  cvt_x<<<(MTOT * DD) / (256 * 4), 256, 0, stream>>>((const float4*)x, xh);
  transpose_cvt<<<dim3(3 * DD / 32, DD / 32), 256, 0, stream>>>(w_qkv, wqkvT, DD, 3 * DD);
  transpose_cvt<<<dim3(DD / 32, DD / 32), 256, 0, stream>>>(w_out, woutT, DD, DD);

  // 2) QKV GEMM: 256 mtiles * 48 ntiles = 12288 waves / 8 = 1536 blocks
  qkv_gemm<<<1536, 256, 0, stream>>>(xh, wqkvT, qb, kbuf, vt);

  // 3) attention: 4*16*64 = 4096 waves / 8 = 512 blocks
  attn_kernel<<<512, 256, 0, stream>>>(qb, kbuf, vt, attn);

  // 4) output projection: 256 * 16 = 4096 waves / 8 = 512 blocks
  out_gemm<<<512, 256, 0, stream>>>(attn, woutT, out);
}
